// text2image_attention_transform_790273983113
// MI455X (gfx1250) — matrix operand order repaired and checked by reading, not compiled, not run on hardware
//
#include <hip/hip_runtime.h>
#include <hip/hip_bf16.h>

// ---------------------------------------------------------------------------
// MI455X (gfx1250) implementation notes:
//  * memory-bound workload (~1.05 GB traffic -> ~45us floor at 23.3 TB/s)
//  * all GEMMs use v_wmma_f32_16x16x32_f16 (wave32 WMMA, f32 accumulate)
//  * attention is fused flash-style: scores never hit HBM
//  * K/V tile staging uses the Tensor Data Mover (tensor_load_to_lds) when the
//    toolchain exposes the builtin; TENSORcnt-synchronized
// ---------------------------------------------------------------------------

typedef __attribute__((ext_vector_type(16))) _Float16     v16h;
typedef __attribute__((ext_vector_type(8)))  float        v8f;
typedef __attribute__((ext_vector_type(4)))  unsigned int u32x4;
typedef __attribute__((ext_vector_type(4)))  unsigned int tdm_g0_t;
typedef __attribute__((ext_vector_type(8)))  int          tdm_g1_t;
typedef __attribute__((ext_vector_type(4)))  int          tdm_g2_t;

#if defined(__gfx1250__) && __has_builtin(__builtin_amdgcn_tensor_load_to_lds) && \
    __has_builtin(__builtin_amdgcn_s_wait_tensorcnt)
#define ATTN_USE_TDM 1
#else
#define ATTN_USE_TDM 0
#endif

union V16 {            // one WMMA f16 operand (16 halfs per lane)
  v16h     v;
  u32x4    u[2];       // two 16-byte chunks
  _Float16 e[16];
};

__device__ __forceinline__ v8f wmma_f16(const V16& a, const V16& b, v8f c) {
  // D = A(16x32 f16) * B(32x16 f16) + C(16x16 f32)
  return __builtin_amdgcn_wmma_f32_16x16x32_f16(
      /*neg_a=*/false, a.v, /*neg_b=*/false, b.v,
      /*c_mod=*/(short)0, c, /*reuse_a=*/false, /*reuse_b=*/false);
}

// ---------------------------------------------------------------------------
// 1) mean over Lv=16, pre-scaled by 1/sqrt(D) so scores = Q*K^T needs no scale
// ---------------------------------------------------------------------------
__global__ void mean_vision_kernel(const float* __restrict__ vp,
                                   _Float16* __restrict__ mv) {
  const int n = blockIdx.x;
  const float sc = (1.0f / 16.0f) * rsqrtf(768.0f);
  const float* base = vp + (size_t)n * 16 * 768;
  for (int d = threadIdx.x; d < 768; d += 256) {
    float s = 0.f;
#pragma unroll
    for (int l = 0; l < 16; ++l) s += base[l * 768 + d];
    mv[(size_t)n * 768 + d] = (_Float16)(s * sc);
  }
}

// ---------------------------------------------------------------------------
// 2) mean over Lt=77
// ---------------------------------------------------------------------------
__global__ void mean_text_kernel(const float* __restrict__ tp,
                                 _Float16* __restrict__ mt) {
  const int m = blockIdx.x;
  const float* base = tp + (size_t)m * 77 * 512;
  for (int d = threadIdx.x; d < 512; d += 256) {
    float s = 0.f;
    for (int l = 0; l < 77; ++l) s += base[l * 512 + d];
    mt[(size_t)m * 512 + d] = (_Float16)(s * (1.0f / 77.0f));
  }
}

// ---------------------------------------------------------------------------
// 3) f32 -> f16 convert (weights)
// ---------------------------------------------------------------------------
__global__ void cvt_kernel(const float* __restrict__ in,
                           _Float16* __restrict__ out, int n) {
  int i = blockIdx.x * 256 + threadIdx.x;
  if (i < n) out[i] = (_Float16)in[i];
}

// ---------------------------------------------------------------------------
// 4) Generic WMMA GEMM:  C[M,N] = A[M,K] * B[N,K]^T + bias[N]   (Linear layout)
//    One wave per block; wave computes a 16x64 output tile.
// ---------------------------------------------------------------------------
__global__ __launch_bounds__(32)
void gemm_rrt_f16(const _Float16* __restrict__ A, const _Float16* __restrict__ B,
                  const float* __restrict__ bias,
                  _Float16* __restrict__ Cf16, float* __restrict__ Cf32,
                  int M, int N, int K, int relu) {
  const int lane = threadIdx.x;
  const int row  = lane & 15;   // A row / B column within tile
  const int hi   = lane >> 4;   // K half-select per WMMA layout
  const int m0   = blockIdx.x * 16;
  const int n0   = blockIdx.y * 64;

  v8f zero = {};
  v8f acc[4] = {zero, zero, zero, zero};

  for (int k0 = 0; k0 < K; k0 += 32) {
    V16 a;
    const _Float16* ap = A + (size_t)(m0 + row) * K + k0 + hi * 16;
    a.u[0] = *(const u32x4*)(ap);
    a.u[1] = *(const u32x4*)(ap + 8);
#pragma unroll
    for (int nt = 0; nt < 4; ++nt) {
      V16 b;
      const _Float16* bp = B + (size_t)(n0 + nt * 16 + row) * K + k0 + hi * 16;
      b.u[0] = *(const u32x4*)(bp);
      b.u[1] = *(const u32x4*)(bp + 8);
      acc[nt] = wmma_f16(a, b, acc[nt]);
    }
  }

#pragma unroll
  for (int nt = 0; nt < 4; ++nt) {
    const int col = n0 + nt * 16 + row;
    const float bia = bias ? bias[col] : 0.f;
#pragma unroll
    for (int j = 0; j < 8; ++j) {
      const int r = m0 + j + hi * 8;  // C layout: VGPR j -> rows j / j+8
      float v = acc[nt][j] + bia;
      if (relu) v = fmaxf(v, 0.f);
      if (Cf16) Cf16[(size_t)r * N + col] = (_Float16)v;
      else      Cf32[(size_t)r * N + col] = v;
    }
  }
}

// ---------------------------------------------------------------------------
// 5) Fused attention: agg = softmax(Q*T^T) * T   (flash-style, online softmax)
//    grid = (N/16, D/192); 1 wave per block owns 16 rows x 192-col D-chunk.
//    K/V rows are staged into LDS by the Tensor Data Mover when available.
// ---------------------------------------------------------------------------
__global__ __launch_bounds__(32)
void attn_softmax_agg_kernel(const _Float16* __restrict__ Q,   // [4096,768] pre-scaled
                             const _Float16* __restrict__ T,   // [4096,768]
                             float* __restrict__ agg) {        // [4096,768]
  constexpr int Dm  = 768;
#if ATTN_USE_TDM
  constexpr int LDT = Dm;                         // TDM writes dense rows
#else
  constexpr int LDT = Dm + 8;                     // pad to dodge bank conflicts
#endif
  __shared__ _Float16 lds_t[32 * LDT];            // staged K/V rows (~48-50 KB)
  __shared__ _Float16 lds_p[16 * 32];             // P tile, C->A layout bounce

  const int lane  = threadIdx.x;
  const int row   = lane & 15;
  const int hi    = lane >> 4;
  const int n0    = blockIdx.x * 16;
  const int dbase = blockIdx.y * 192;

  v8f zero = {};
  v8f acc[12];
#pragma unroll
  for (int i = 0; i < 12; ++i) acc[i] = zero;
  float runmax[8], runsum[8];
#pragma unroll
  for (int j = 0; j < 8; ++j) { runmax[j] = -1e30f; runsum[j] = 0.f; }

  const _Float16* qbase = Q + (size_t)n0 * Dm;

  for (int m0 = 0; m0 < 4096; m0 += 32) {
#if ATTN_USE_TDM
    // -------- TDM staging: one 2-D tile DMA (32 rows x 768 f16) -> LDS -----
    {
      // previous iteration's LDS reads must retire before TDM overwrites LDS
      asm volatile("s_wait_dscnt 0" ::: "memory");
      const unsigned long long ga =
          (unsigned long long)(size_t)(const void*)T +
          (unsigned long long)m0 * (Dm * 2);          // tile start (bytes)
      const unsigned lds_off = (unsigned)(size_t)(void*)&lds_t[0];
      tdm_g0_t g0;
      g0.x = 1u;                                      // count=1 (valid D#)
      g0.y = lds_off;                                 // lds_addr
      g0.z = (unsigned)ga;                            // global_addr[31:0]
      g0.w = (unsigned)((ga >> 32) & 0x01FFFFFFull)   // global_addr[56:32]
             | (2u << 30);                            // type=2 ("image")
      tdm_g1_t g1;
      g1[0] = 0x00010000;                             // data_size=1 (2B), no mcast
      g1[1] = (int)(((unsigned)Dm & 0xFFFFu) << 16);  // tensor_dim0[15:0]
      g1[2] = (int)(((unsigned)Dm >> 16) | ((4096u & 0xFFFFu) << 16)); // d0 hi|d1 lo
      g1[3] = (int)((4096u >> 16) | ((unsigned)Dm << 16)); // d1 hi | tile_dim0=768
      g1[4] = 32;                                     // tile_dim1=32, tile_dim2=0
      g1[5] = Dm;                                     // tensor_dim0_stride lo32
      g1[6] = 0;                                      // stride hi | dim1_stride lo
      g1[7] = 0;
      tdm_g2_t gz;  gz[0] = gz[1] = gz[2] = gz[3] = 0;   // 2-D tensor: unused
#if __clang_major__ >= 23
      tdm_g1_t gz8; gz8[0]=gz8[1]=gz8[2]=gz8[3]=gz8[4]=gz8[5]=gz8[6]=gz8[7]=0;
      __builtin_amdgcn_tensor_load_to_lds(g0, g1, gz, gz, gz8, 0);
#else
      __builtin_amdgcn_tensor_load_to_lds(g0, g1, gz, gz, 0);
#endif
      if (m0 + 32 < 4096)
        __builtin_prefetch(T + (size_t)(m0 + 32) * Dm, 0, 0);
      __builtin_amdgcn_s_wait_tensorcnt(0);           // tile resident in LDS
      asm volatile("" ::: "memory");
    }
#else
    __syncthreads();                                  // lds_t reuse fence
    {
      // stage 32 rows of T (b128 global loads -> b128 LDS stores)
      const u32x4* src = (const u32x4*)(T + (size_t)m0 * Dm);
      for (int i = lane; i < 32 * (Dm / 8); i += 32) {
        const int r = i / (Dm / 8), c = i % (Dm / 8);
        *(u32x4*)&lds_t[r * LDT + c * 8] = src[i];
      }
      if (m0 + 32 < 4096)                             // gfx1250 prefetch path
        __builtin_prefetch(T + (size_t)(m0 + 32) * Dm, 0, 0);
    }
    __syncthreads();
#endif

    // ---- scores: S(16x32) = Q_tile(16x768) * T_tile(32x768)^T -------------
    v8f s0 = {}, s1 = {};
#pragma unroll
    for (int kk = 0; kk < Dm / 32; ++kk) {
      V16 a, b0, b1;
      const _Float16* ap = qbase + (size_t)row * Dm + kk * 32 + hi * 16;
      a.u[0] = *(const u32x4*)(ap);
      a.u[1] = *(const u32x4*)(ap + 8);
      const _Float16* b0p = &lds_t[row * LDT + kk * 32 + hi * 16];
      b0.u[0] = *(const u32x4*)(b0p);
      b0.u[1] = *(const u32x4*)(b0p + 8);
      const _Float16* b1p = &lds_t[(row + 16) * LDT + kk * 32 + hi * 16];
      b1.u[0] = *(const u32x4*)(b1p);
      b1.u[1] = *(const u32x4*)(b1p + 8);
      s0 = wmma_f16(a, b0, s0);
      s1 = wmma_f16(a, b1, s1);
    }

    // ---- online softmax (row reductions across 16-lane halves) ------------
#pragma unroll
    for (int j = 0; j < 8; ++j) {
      float a = s0[j], b = s1[j];
      float m = fmaxf(a, b);
      m = fmaxf(m, __shfl_xor(m, 1, 32));
      m = fmaxf(m, __shfl_xor(m, 2, 32));
      m = fmaxf(m, __shfl_xor(m, 4, 32));
      m = fmaxf(m, __shfl_xor(m, 8, 32));
      const float nm  = fmaxf(runmax[j], m);
      const float fac = __expf(runmax[j] - nm);
      runmax[j] = nm;
      const float p0 = __expf(a - nm);
      const float p1 = __expf(b - nm);
      float ts = p0 + p1;
      ts += __shfl_xor(ts, 1, 32);
      ts += __shfl_xor(ts, 2, 32);
      ts += __shfl_xor(ts, 4, 32);
      ts += __shfl_xor(ts, 8, 32);
      runsum[j] = runsum[j] * fac + ts;
#pragma unroll
      for (int t = 0; t < 12; ++t) acc[t][j] *= fac;   // rescale running agg
      const int pr = j + hi * 8;                        // C-layout row
      lds_p[pr * 32 + row]      = (_Float16)p0;
      lds_p[pr * 32 + row + 16] = (_Float16)p1;
    }
    __syncthreads();

    // ---- agg += P(16x32) * V(32x192-chunk) --------------------------------
    V16 pa;
    const _Float16* pp = &lds_p[row * 32 + hi * 16];    // A-layout read of P
    pa.u[0] = *(const u32x4*)(pp);
    pa.u[1] = *(const u32x4*)(pp + 8);
#pragma unroll
    for (int nt = 0; nt < 12; ++nt) {
      V16 bv;                                           // B 32x16: col=lane, K strided
#pragma unroll
      for (int k = 0; k < 16; ++k)
        bv.e[k] = lds_t[(hi * 16 + k) * LDT + dbase + nt * 16 + row];
      acc[nt] = wmma_f16(pa, bv, acc[nt]);
    }
  }

  // ---- normalize by softmax denominator and store -------------------------
#pragma unroll
  for (int j = 0; j < 8; ++j) {
    const float inv = 1.0f / runsum[j];
    const int r = n0 + j + hi * 8;
#pragma unroll
    for (int nt = 0; nt < 12; ++nt)
      agg[(size_t)r * Dm + dbase + nt * 16 + row] = acc[nt][j] * inv;
  }
}

// ---------------------------------------------------------------------------
// 6) LayerNorm over D=768, output f16 for the MLP GEMMs
// ---------------------------------------------------------------------------
__global__ void layernorm_kernel(const float* __restrict__ x,
                                 const float* __restrict__ g,
                                 const float* __restrict__ b,
                                 _Float16* __restrict__ out) {
  __shared__ float red[256];
  const int n = blockIdx.x, tid = threadIdx.x;
  const float* xr = x + (size_t)n * 768;

  float s = 0.f;
  for (int d = tid; d < 768; d += 256) s += xr[d];
  red[tid] = s; __syncthreads();
  for (int off = 128; off > 0; off >>= 1) {
    if (tid < off) red[tid] += red[tid + off];
    __syncthreads();
  }
  const float mu = red[0] * (1.0f / 768.0f);
  __syncthreads();

  float v = 0.f;
  for (int d = tid; d < 768; d += 256) { float t = xr[d] - mu; v += t * t; }
  red[tid] = v; __syncthreads();
  for (int off = 128; off > 0; off >>= 1) {
    if (tid < off) red[tid] += red[tid + off];
    __syncthreads();
  }
  const float rstd = rsqrtf(red[0] * (1.0f / 768.0f) + 1e-5f);

  for (int d = tid; d < 768; d += 256)
    out[(size_t)n * 768 + d] = (_Float16)((xr[d] - mu) * rstd * g[d] + b[d]);
}

// ---------------------------------------------------------------------------
// 7) Broadcast h[N,768] across Lv=16 -> out[N,16,768] (float4 stores, 201 MB)
// ---------------------------------------------------------------------------
__global__ void broadcast_kernel(const float* __restrict__ h,
                                 float* __restrict__ out) {
  const int idx = blockIdx.x * 256 + threadIdx.x;     // over N * 192 float4s
  if (idx >= 4096 * 192) return;
  const int n = idx / 192, c = idx % 192;
  const float4 v = ((const float4*)h)[idx];
  float4* o = (float4*)out + (size_t)n * 16 * 192 + c;
#pragma unroll
  for (int l = 0; l < 16; ++l) o[l * 192] = v;
}

// ---------------------------------------------------------------------------
// host-side orchestration
// ---------------------------------------------------------------------------
extern "C" void kernel_launch(void* const* d_in, const int* in_sizes, int n_in,
                              void* d_out, int out_size, void* d_ws, size_t ws_size,
                              hipStream_t stream) {
  (void)in_sizes; (void)n_in; (void)out_size; (void)ws_size;
  const float* vp   = (const float*)d_in[0];
  const float* tp   = (const float*)d_in[1];
  const float* fc_w = (const float*)d_in[2];
  const float* fc_b = (const float*)d_in[3];
  const float* ln_g = (const float*)d_in[4];
  const float* ln_b = (const float*)d_in[5];
  const float* w1   = (const float*)d_in[6];
  const float* b1   = (const float*)d_in[7];
  const float* w2   = (const float*)d_in[8];
  const float* b2   = (const float*)d_in[9];
  float* out = (float*)d_out;

  char* wsp = (char*)d_ws;
  auto alloc = [&](size_t bytes) {
    char* p = wsp;
    wsp += (bytes + 255) & ~(size_t)255;
    return p;
  };
  _Float16* mv   = (_Float16*)alloc(4096ull * 768 * 2);  // mean_v (f16, pre-scaled)
  _Float16* mt   = (_Float16*)alloc(4096ull * 512 * 2);  // mean_t (f16)
  _Float16* fcwh = (_Float16*)alloc(768ull * 512 * 2);   // fc_w   (f16)
  _Float16* th   = (_Float16*)alloc(4096ull * 768 * 2);  // t      (f16)
  float*    agg  = (float*)   alloc(4096ull * 768 * 4);  // softmax(QK)*V
  _Float16* hh   = (_Float16*)alloc(4096ull * 768 * 2);  // LN(agg) (f16)
  _Float16* w1h  = (_Float16*)alloc(192ull * 768 * 2);
  _Float16* gh   = (_Float16*)alloc(4096ull * 192 * 2);  // relu(h@W1^T)
  _Float16* w2h  = (_Float16*)alloc(768ull * 192 * 2);
  float*    h2   = (float*)   alloc(4096ull * 768 * 4);  // final rows (f32)

  mean_vision_kernel<<<4096, 256, 0, stream>>>(vp, mv);
  mean_text_kernel  <<<4096, 256, 0, stream>>>(tp, mt);
  cvt_kernel<<<(768 * 512 + 255) / 256, 256, 0, stream>>>(fc_w, fcwh, 768 * 512);
  cvt_kernel<<<(192 * 768 + 255) / 256, 256, 0, stream>>>(w1, w1h, 192 * 768);
  cvt_kernel<<<(768 * 192 + 255) / 256, 256, 0, stream>>>(w2, w2h, 768 * 192);

  // t = mean_t @ fc_w^T + fc_b        [4096 x 768], K = 512
  gemm_rrt_f16<<<dim3(4096 / 16, 768 / 64), 32, 0, stream>>>(
      mt, fcwh, fc_b, th, nullptr, 4096, 768, 512, 0);

  // agg = softmax(mean_v @ t^T / sqrt(D)) @ t    (scale folded into mv)
  attn_softmax_agg_kernel<<<dim3(4096 / 16, 768 / 192), 32, 0, stream>>>(mv, th, agg);

  layernorm_kernel<<<4096, 256, 0, stream>>>(agg, ln_g, ln_b, hh);

  // h1 = relu(h @ W1^T + b1)          [4096 x 192], K = 768
  gemm_rrt_f16<<<dim3(4096 / 16, 192 / 64), 32, 0, stream>>>(
      hh, w1h, b1, gh, nullptr, 4096, 192, 768, 1);

  // h2 = h1 @ W2^T + b2               [4096 x 768], K = 192 (f32 out)
  gemm_rrt_f16<<<dim3(4096 / 16, 768 / 64), 32, 0, stream>>>(
      gh, w2h, b2, nullptr, h2, 4096, 768, 192, 0);

  broadcast_kernel<<<(4096 * 192 + 255) / 256, 256, 0, stream>>>(h2, out);
}
